// MultiHeadAttention_17686675325088
// MI455X (gfx1250) — matrix operand order repaired and checked by reading, not compiled, not run on hardware
//
#include <hip/hip_runtime.h>

typedef _Float16 h8   __attribute__((ext_vector_type(8)));
typedef _Float16 h16  __attribute__((ext_vector_type(16)));
typedef float    f8   __attribute__((ext_vector_type(8)));

#define SEQ 2048
#define HIDC 1024
#define NBATCH 4
#define NHEAD 16
#define HD 64

// Build a 16-bit A-matrix fragment (16x32, MxK) for one lane:
// halves 0..7  <- p[0..7]   (K = kb .. kb+7)
// halves 8..15 <- p[8..15]  (K = kb+8 .. kb+15)
__device__ __forceinline__ h16 load_a16(const _Float16* p) {
  h8 lo = *(const h8*)(p);
  h8 hi = *(const h8*)(p + 8);
  return __builtin_shufflevector(lo, hi, 0,1,2,3,4,5,6,7,8,9,10,11,12,13,14,15);
}

__global__ void f32_to_f16_kernel(const float* __restrict__ x,
                                  _Float16* __restrict__ y, int n) {
  int i = blockIdx.x * blockDim.x + threadIdx.x;
  if (i < n) y[i] = (_Float16)x[i];
}

// Vh: [B, SEQ, HID] f16  ->  Vt: [B, H, HD, SEQ] f16 (coalesced writes)
__global__ void transpose_v_kernel(const _Float16* __restrict__ Vh,
                                   _Float16* __restrict__ Vt) {
  int idx = blockIdx.x * 256 + threadIdx.x;
  int n = idx & (SEQ - 1);
  int d = (idx >> 11) & (HD - 1);
  int h = (idx >> 17) & (NHEAD - 1);
  int b = idx >> 21;
  Vt[idx] = Vh[((size_t)(b * SEQ + n)) * HIDC + h * HD + d];
}

// Y[m,n] = sum_k A[m,k] * W[n,k] + bias[n]
// A: [M,K] f16 row-major, W: [N,K] f16 row-major (nn.Linear weight layout).
// Block: 256 threads = 8 waves (4 along M x 2 along N); wave tile 64x64
// (4x4 accumulators, 16 WMMA per 32-wide k-step).
template<bool HALF_OUT>
__global__ __launch_bounds__(256) void gemm_nt_kernel(
    const _Float16* __restrict__ A, const _Float16* __restrict__ W,
    const float* __restrict__ bias, void* __restrict__ Y,
    int M, int N, int K)
{
  const int lane = threadIdx.x & 31;
  const int wave = threadIdx.x >> 5;
  const int m0 = blockIdx.x * 256 + (wave & 3) * 64;
  const int n0 = blockIdx.y * 128 + (wave >> 2) * 64;

  const int am = lane & 15, akb = (lane >> 4) * 16;  // A frag: row, K-base
  const int bn = lane & 15, bkb = (lane >> 4) * 16;  // B frag: col, K-base

  f8 acc[4][4] = {};

  for (int k0 = 0; k0 < K; k0 += 32) {
    if (k0 + 64 < K) {  // pull next-next k-step toward L0 (global_prefetch_b8)
      __builtin_prefetch(W + (size_t)(n0 + bn) * K + k0 + 64 + bkb, 0, 1);
      __builtin_prefetch(A + (size_t)(m0 + am) * K + k0 + 64 + akb, 0, 1);
    }
    h16 bfrag[4];
#pragma unroll
    for (int j = 0; j < 4; ++j)
      bfrag[j] = *(const h16*)(W + (size_t)(n0 + j * 16 + bn) * K + k0 + bkb);
#pragma unroll
    for (int i = 0; i < 4; ++i) {
      h16 afrag = load_a16(A + (size_t)(m0 + i * 16 + am) * K + k0 + akb);
#pragma unroll
      for (int j = 0; j < 4; ++j)
        acc[i][j] = __builtin_amdgcn_wmma_f32_16x16x32_f16(
            false, afrag, false, bfrag[j], (short)0, acc[i][j], false, false);
    }
  }

  const int cn = lane & 15, cmb = (lane >> 4) * 8;
#pragma unroll
  for (int i = 0; i < 4; ++i) {
#pragma unroll
    for (int j = 0; j < 4; ++j) {
      const int col = n0 + j * 16 + cn;
      const float bval = bias[col];
#pragma unroll
      for (int r = 0; r < 8; ++r) {
        const int row = m0 + i * 16 + cmb + r;
        const float v = acc[i][j][r] + bval;
        if (HALF_OUT) ((_Float16*)Y)[(size_t)row * N + col] = (_Float16)v;
        else          ((float*)Y)[(size_t)row * N + col]    = v;
      }
    }
  }
}

// Flash attention: block = 4 waves x 16 q-rows = 64 q rows per block.
// Q,K: [B, SEQ, HID] f16 (head h at columns h*64..); V: [B,H,HD,SEQ] f16.
// O (f16) written back to [B, SEQ, HID] layout.
__global__ __launch_bounds__(128) void attn_kernel(
    const _Float16* __restrict__ Qh, const _Float16* __restrict__ Kh,
    const _Float16* __restrict__ Vt, _Float16* __restrict__ Oh)
{
  const int lane = threadIdx.x & 31;
  const int wave = threadIdx.x >> 5;
  const int h = blockIdx.y, b = blockIdx.z;
  const int q0 = blockIdx.x * 64 + wave * 16;

  const _Float16* Q = Qh + (size_t)b * SEQ * HIDC + h * HD;
  const _Float16* K = Kh + (size_t)b * SEQ * HIDC + h * HD;
  const _Float16* V = Vt + ((size_t)(b * NHEAD + h)) * HD * SEQ;

  const int am = lane & 15, akb = (lane >> 4) * 16;
  const int bn = lane & 15, bkb = (lane >> 4) * 16;
  const int cn = lane & 15, cmb = (lane >> 4) * 8;

  __shared__ alignas(32) _Float16 pbuf[4][16][32];  // per-wave P staging

  // Preload this wave's Q tile (16 x 64) as two A fragments.
  h16 qf[2];
#pragma unroll
  for (int kk = 0; kk < 2; ++kk)
    qf[kk] = load_a16(Q + (size_t)(q0 + am) * HIDC + kk * 32 + akb);

  f8 oacc[4] = {};
  float mrow[8], lrow[8];
#pragma unroll
  for (int r = 0; r < 8; ++r) { mrow[r] = -1.0e30f; lrow[r] = 0.0f; }

  // softmax(energy / sqrt(HID)): fold 1/32 and log2(e) into exp2 argument.
  const float sc = 1.44269504f / 32.0f;

  for (int k0 = 0; k0 < SEQ; k0 += 32) {
    if (k0 + 32 < SEQ) {  // prefetch next K rows / V columns tile
      __builtin_prefetch(K + (size_t)(k0 + 32 + bn) * HIDC + bkb, 0, 1);
      __builtin_prefetch(V + (size_t)bn * SEQ + k0 + 32 + bkb, 0, 1);
    }
    // S tiles: 16 q-rows x 32 k-cols (two 16x16 WMMA results)
    f8 s[2];
#pragma unroll
    for (int j = 0; j < 2; ++j) {
      f8 sj = {};
#pragma unroll
      for (int kk = 0; kk < 2; ++kk) {
        h16 kf = *(const h16*)(K + (size_t)(k0 + j * 16 + bn) * HIDC + kk * 32 + bkb);
        sj = __builtin_amdgcn_wmma_f32_16x16x32_f16(
            false, qf[kk], false, kf, (short)0, sj, false, false);
      }
      s[j] = sj;
    }

    // Online softmax: C-layout row r lives across the 16 lanes of a half-group.
#pragma unroll
    for (int r = 0; r < 8; ++r) {
      float vmax = fmaxf(s[0][r], s[1][r]);
#pragma unroll
      for (int off = 1; off < 16; off <<= 1)
        vmax = fmaxf(vmax, __shfl_xor(vmax, off, 32));
      const float mnew  = fmaxf(mrow[r], vmax);
      const float alpha = exp2f((mrow[r] - mnew) * sc);
      mrow[r] = mnew;
      const float p0 = exp2f((s[0][r] - mnew) * sc);
      const float p1 = exp2f((s[1][r] - mnew) * sc);
      float ps = p0 + p1;
#pragma unroll
      for (int off = 1; off < 16; off <<= 1)
        ps += __shfl_xor(ps, off, 32);
      lrow[r] = lrow[r] * alpha + ps;
#pragma unroll
      for (int t = 0; t < 4; ++t) oacc[t][r] *= alpha;
      pbuf[wave][cmb + r][cn]      = (_Float16)p0;
      pbuf[wave][cmb + r][cn + 16] = (_Float16)p1;
    }

    // Intra-wave LDS RAW: C-layout -> A-layout through LDS.
    asm volatile("s_wait_dscnt 0" ::: "memory");
    h16 pf = load_a16(&pbuf[wave][am][akb]);

    // O += P(16x32) @ V(32x64): B frag is a contiguous row read of Vt.
#pragma unroll
    for (int t = 0; t < 4; ++t) {
      h16 vf = *(const h16*)(V + (size_t)(t * 16 + bn) * SEQ + k0 + bkb);
      oacc[t] = __builtin_amdgcn_wmma_f32_16x16x32_f16(
          false, pf, false, vf, (short)0, oacc[t], false, false);
    }
  }

#pragma unroll
  for (int t = 0; t < 4; ++t) {
#pragma unroll
    for (int r = 0; r < 8; ++r) {
      const float v = oacc[t][r] / lrow[r];
      Oh[(size_t)(b * SEQ + q0 + cmb + r) * HIDC + h * HD + t * 16 + cn] = (_Float16)v;
    }
  }
}

extern "C" void kernel_launch(void* const* d_in, const int* in_sizes, int n_in,
                              void* d_out, int out_size, void* d_ws, size_t ws_size,
                              hipStream_t stream) {
  const float* x  = (const float*)d_in[0];
  const float* Wq = (const float*)d_in[1];
  const float* bq = (const float*)d_in[2];
  const float* Wk = (const float*)d_in[3];
  const float* bk = (const float*)d_in[4];
  const float* Wv = (const float*)d_in[5];
  const float* bv = (const float*)d_in[6];
  const float* Wo = (const float*)d_in[7];
  const float* bo = (const float*)d_in[8];

  const size_t NX = (size_t)NBATCH * SEQ * HIDC;  // 8388608 activations
  const size_t NW = (size_t)HIDC * HIDC;          // 1048576 per weight

  _Float16* ws  = (_Float16*)d_ws;
  _Float16* xh  = ws;           // x in f16 (reused later for attention output)
  _Float16* Wqh = xh  + NX;
  _Float16* Wkh = Wqh + NW;
  _Float16* Wvh = Wkh + NW;
  _Float16* Woh = Wvh + NW;
  _Float16* Qh  = Woh + NW;
  _Float16* Kh  = Qh  + NX;
  _Float16* Vh  = Kh  + NX;
  _Float16* Vtr = Vh  + NX;
  _Float16* Ohb = xh;           // reuse: x no longer needed after V projection

  const int M = NBATCH * SEQ;   // 8192

  f32_to_f16_kernel<<<(int)(NX / 256), 256, 0, stream>>>(x,  xh,  (int)NX);
  f32_to_f16_kernel<<<(int)(NW / 256), 256, 0, stream>>>(Wq, Wqh, (int)NW);
  f32_to_f16_kernel<<<(int)(NW / 256), 256, 0, stream>>>(Wk, Wkh, (int)NW);
  f32_to_f16_kernel<<<(int)(NW / 256), 256, 0, stream>>>(Wv, Wvh, (int)NW);
  f32_to_f16_kernel<<<(int)(NW / 256), 256, 0, stream>>>(Wo, Woh, (int)NW);

  dim3 gg(M / 256, HIDC / 128);
  gemm_nt_kernel<true ><<<gg, 256, 0, stream>>>(xh, Wqh, bq, Qh, M, HIDC, HIDC);
  gemm_nt_kernel<true ><<<gg, 256, 0, stream>>>(xh, Wkh, bk, Kh, M, HIDC, HIDC);
  gemm_nt_kernel<true ><<<gg, 256, 0, stream>>>(xh, Wvh, bv, Vh, M, HIDC, HIDC);

  transpose_v_kernel<<<(int)(NX / 256), 256, 0, stream>>>(Vh, Vtr);

  attn_kernel<<<dim3(SEQ / 64, NHEAD, NBATCH), 128, 0, stream>>>(Qh, Kh, Vtr, Ohb);

  gemm_nt_kernel<false><<<gg, 256, 0, stream>>>(Ohb, Woh, bo, d_out, M, HIDC, HIDC);
}
